// PHLoss_63299228009122
// MI455X (gfx1250) — compile-verified, matrix operand order checked by reading, not run on hardware
//
#include <hip/hip_runtime.h>
#include <math.h>

#define PH_H 64
#define PH_W 64
#define PH_N 4096      // 64*64
#define NSLICE 8       // B*C
#define NTHR 256       // 8 wave32s

typedef __attribute__((ext_vector_type(2)))  float    v2f;
typedef __attribute__((ext_vector_type(8)))  float    v8f;
typedef __attribute__((ext_vector_type(16))) _Float16 v16h;

__device__ __forceinline__ int uf_find(unsigned short* par, int x) {
  // path-halving find, single-thread use only
  while ((int)par[x] != x) {
    int gp = (int)par[(int)par[x]];
    par[x] = (unsigned short)gp;
    x = gp;
  }
  return x;
}

__global__ __launch_bounds__(NTHR, 1)
void ph_slice_kernel(const float* __restrict__ pred,
                     const float* __restrict__ target,
                     float* __restrict__ losses) {
  __shared__ float          s_fval[PH_N];   // filtration value per pixel
  __shared__ float          s_life[PH_N];   // collected lifetimes (then sorted desc)
  __shared__ unsigned short s_idx [PH_N];   // argsort order
  __shared__ unsigned short s_rank[PH_N];   // insertion rank per pixel
  __shared__ unsigned short s_par [PH_N];   // union-find parent
  __shared__ float          s_red [NTHR];   // reduction scratch
  __shared__ float          s_scal[2];      // [0]=min filtration, [1]=max
  __shared__ int            s_cnt;

  const int tid   = threadIdx.x;
  const int slice = blockIdx.x;

  float lpreg[16];                          // pred diagram cached in registers
  #pragma unroll
  for (int r = 0; r < 16; ++r) lpreg[r] = 0.0f;

  for (int pass = 0; pass < 2; ++pass) {
    const float* src = (pass == 0 ? pred : target) + slice * PH_N;
    __builtin_prefetch(src + tid, 0, 1);    // -> global_prefetch_b8

    // ---- filtration ----
    for (int e = tid; e < PH_N; e += NTHR) {
      float x = src[e];
      float f = (pass == 0) ? (1.0f / (1.0f + expf(x)))   // 1 - sigmoid(x)
                            : (1.0f - x);
      s_fval[e] = f;
      s_idx[e]  = (unsigned short)e;
    }
    __syncthreads();

    // ---- min / max for essential class ----
    {
      float lmin = 3.4e38f, lmax = -3.4e38f;
      for (int e = tid; e < PH_N; e += NTHR) {
        float v = s_fval[e];
        lmin = fminf(lmin, v);
        lmax = fmaxf(lmax, v);
      }
      s_red[tid] = lmax;
      __syncthreads();
      for (int off = NTHR / 2; off > 0; off >>= 1) {
        if (tid < off) s_red[tid] = fmaxf(s_red[tid], s_red[tid + off]);
        __syncthreads();
      }
      if (tid == 0) s_scal[1] = s_red[0];
      __syncthreads();
      s_red[tid] = lmin;
      __syncthreads();
      for (int off = NTHR / 2; off > 0; off >>= 1) {
        if (tid < off) s_red[tid] = fminf(s_red[tid], s_red[tid + off]);
        __syncthreads();
      }
      if (tid == 0) s_scal[0] = s_red[0];
      __syncthreads();
    }

    // ---- bitonic argsort of pixels by filtration value (ascending) ----
    for (int k = 2; k <= PH_N; k <<= 1) {
      for (int j = k >> 1; j > 0; j >>= 1) {
        __syncthreads();
        for (int t = tid; t < PH_N; t += NTHR) {
          int ixj = t ^ j;
          if (ixj > t) {
            bool up = ((t & k) == 0);
            unsigned short ia = s_idx[t], ib = s_idx[ixj];
            float ka = s_fval[ia], kb = s_fval[ib];
            if ((ka > kb) == up) { s_idx[t] = ib; s_idx[ixj] = ia; }
          }
        }
      }
    }
    __syncthreads();

    // ---- rank + union-find init ----
    for (int i = tid; i < PH_N; i += NTHR) {
      s_rank[s_idx[i]] = (unsigned short)i;
      s_par[i]         = (unsigned short)i;
    }
    __syncthreads();

    // ---- sequential 0-dim persistence scan (elder rule) ----
    // birth(root) == f(root): the surviving root is always the lower-birth one.
    if (tid == 0) {
      int cnt = 0;
      for (int i = 0; i < PH_N; ++i) {
        const int   p  = (int)s_idx[i];
        const float v  = s_fval[p];
        const int   pi = p >> 6;
        const int   pj = p & 63;
        #pragma unroll
        for (int k = 0; k < 4; ++k) {
          int q; bool ok;
          if      (k == 0) { q = p - PH_W; ok = (pi > 0);        }
          else if (k == 1) { q = p + PH_W; ok = (pi < PH_H - 1); }
          else if (k == 2) { q = p - 1;    ok = (pj > 0);        }
          else             { q = p + 1;    ok = (pj < PH_W - 1); }
          if (!ok) continue;
          if ((int)s_rank[q] >= i) continue;   // neighbor not inserted yet
          int rp = uf_find(s_par, p);
          int rq = uf_find(s_par, q);
          if (rp == rq) continue;
          float bp = s_fval[rp];
          float bq = s_fval[rq];
          bool p_young = (bp > bq) ||
                         ((bp == bq) && ((int)s_rank[rp] > (int)s_rank[rq]));
          int   young = p_young ? rp : rq;
          int   older = p_young ? rq : rp;
          float life  = v - (p_young ? bp : bq);
          s_par[young] = (unsigned short)older;
          if (life > 0.0f) s_life[cnt++] = life;   // zeros are implicit padding
        }
      }
      s_life[cnt++] = s_scal[1] - s_scal[0];       // essential class
      s_cnt = cnt;                                  // cnt <= 4096 always
    }
    __syncthreads();

    // ---- zero-pad remainder ----
    {
      int cnt = s_cnt;
      for (int e = tid; e < PH_N; e += NTHR)
        if (e >= cnt) s_life[e] = 0.0f;
    }

    // ---- bitonic sort lifetimes descending ----
    for (int k = 2; k <= PH_N; k <<= 1) {
      for (int j = k >> 1; j > 0; j >>= 1) {
        __syncthreads();
        for (int t = tid; t < PH_N; t += NTHR) {
          int ixj = t ^ j;
          if (ixj > t) {
            bool up = ((t & k) == 0);
            float ka = s_life[t], kb = s_life[ixj];
            if ((ka < kb) == up) { s_life[t] = kb; s_life[ixj] = ka; }
          }
        }
      }
    }
    __syncthreads();

    if (pass == 0) {
      // cache pred's sorted diagram in registers
      #pragma unroll
      for (int r = 0; r < 16; ++r) lpreg[r] = s_life[tid + NTHR * r];
    } else {
      // squared-difference partial sums
      float partial = 0.0f;
      #pragma unroll
      for (int r = 0; r < 16; ++r) {
        float d = s_life[tid + NTHR * r] - lpreg[r];
        partial += d * d;
      }
      s_red[tid] = partial;
      __syncthreads();
      if (tid < 64)
        s_red[tid] = s_red[tid] + s_red[tid + 64] + s_red[tid + 128] + s_red[tid + 192];
      __syncthreads();

      if (tid < 32) {   // wave 0 fully active -> EXEC all ones for WMMA
#if __has_builtin(__builtin_amdgcn_wmma_f32_16x16x4_f32)
        // Ones-matrix WMMA reduction: D[m][n] = rowsum_m, so sum(D) = 16 * total.
        int m  = tid & 15;
        int kb = (tid >> 4) * 2;
        v2f a; a.x = s_red[m * 4 + kb]; a.y = s_red[m * 4 + kb + 1];
        v2f b; b.x = 1.0f; b.y = 1.0f;
        v8f c = {};
        c = __builtin_amdgcn_wmma_f32_16x16x4_f32(false, a, false, b,
                                                  (short)0, c, false, false);
#else
        // fallback: codegen-confirmed f16 WMMA, same ones-matrix trick
        v16h a = {};
        a[0] = (_Float16)s_red[tid * 2];
        a[1] = (_Float16)s_red[tid * 2 + 1];
        v16h b;
        #pragma unroll
        for (int z = 0; z < 16; ++z) b[z] = (_Float16)1.0f;
        v8f c = {};
        c = __builtin_amdgcn_wmma_f32_16x16x32_f16(false, a, false, b,
                                                   (short)0, c, false, false);
#endif
        float s = c[0] + c[1] + c[2] + c[3] + c[4] + c[5] + c[6] + c[7];
        // sum all 256 D entries across the wave; each of 16 columns = total
        for (int off = 16; off > 0; off >>= 1) s += __shfl_xor(s, off, 32);
        if (tid == 0) losses[slice] = sqrtf(0.5f * (s * (1.0f / 16.0f)));
      }
    }
    __syncthreads();
  }
}

__global__ void ph_finalize_kernel(const float* __restrict__ losses,
                                   float* __restrict__ out) {
  if (threadIdx.x == 0 && blockIdx.x == 0) {
    float s = 0.0f;
    #pragma unroll
    for (int i = 0; i < NSLICE; ++i) s += losses[i];
    out[0] = s * (1.0f / NSLICE);
  }
}

extern "C" void kernel_launch(void* const* d_in, const int* in_sizes, int n_in,
                              void* d_out, int out_size, void* d_ws, size_t ws_size,
                              hipStream_t stream) {
  const float* pred   = (const float*)d_in[0];    // (4,2,64,64) float32 logits
  const float* target = (const float*)d_in[1];    // (4,2,64,64) float32 {0,1}
  float* losses = (float*)d_ws;                   // 8 per-slice losses
  ph_slice_kernel<<<NSLICE, NTHR, 0, stream>>>(pred, target, losses);
  ph_finalize_kernel<<<1, 32, 0, stream>>>(losses, (float*)d_out);
  (void)in_sizes; (void)n_in; (void)out_size; (void)ws_size;
}
